// DecoderRNN_3255585210774
// MI455X (gfx1250) — matrix operand order repaired and checked by reading, not compiled, not run on hardware
//
#include <hip/hip_runtime.h>
#include <hip/hip_bf16.h>

// Problem constants (from reference)
#define BB 32
#define HH 512
#define EE 512
#define VV 32000
#define TSRC 64
#define LLEN 63          // max_len = T_TGT - 1
#define G3H 1536         // 3*H
#define WPB 4            // waves per block in GEMM kernels

typedef __attribute__((ext_vector_type(16))) __bf16 v16bf;
typedef __attribute__((ext_vector_type(8)))  float  v8f;
typedef __attribute__((ext_vector_type(4)))  int    i4;

static __device__ __forceinline__ unsigned short f2bf(float f) {
    unsigned u = __float_as_uint(f);
    unsigned r = u + 0x7FFFu + ((u >> 16) & 1u);   // round-to-nearest-even
    return (unsigned short)(r >> 16);
}

// ---------------- fp32 -> bf16 convert ----------------
__global__ void cvt_bf16_kernel(const float* __restrict__ src,
                                unsigned short* __restrict__ dst, int n) {
    int i = blockIdx.x * blockDim.x + threadIdx.x;
    if (i < n) dst[i] = f2bf(src[i]);
}

// ---------------- bf16 WMMA GEMM core -----------------------------------------------
// C[32 x N] = A[32 x K] * W[N x K]^T + bias.
// One wave owns one 16-wide column tile and BOTH 16-row m-tiles: each B fragment is
// loaded once and consumed by two back-to-back WMMAs (halves W traffic, the dominant
// bytes). block = (32, WPB): WPB waves -> WPB consecutive n tiles.
struct GemmArgs {
    const unsigned short* A;  int lda;
    const unsigned short* W;  int ldw;
    const float* bias;
    float* C;                 long long ldc;
    int K;
};

static __device__ __forceinline__ void gemm_body(const GemmArgs& p, int nt) {
    const int lane = threadIdx.x;      // 0..31
    const int g    = lane >> 4;        // lane group
    const int r    = lane & 15;

    const unsigned short* Arow0 = p.A + (size_t)r        * p.lda;   // m-tile 0
    const unsigned short* Arow1 = p.A + (size_t)(16 + r) * p.lda;   // m-tile 1
    const unsigned short* Wrow  = p.W + (size_t)(nt * 16 + r) * p.ldw;

    v8f acc0 = {0.f,0.f,0.f,0.f,0.f,0.f,0.f,0.f};
    v8f acc1 = {0.f,0.f,0.f,0.f,0.f,0.f,0.f,0.f};

    for (int k0 = 0; k0 < p.K; k0 += 32) {
        union { v16bf v; i4 q[2]; } a0, a1, b;
        // B 32x16 layout: g=0 -> K {0..7,16..23}; g=1 -> K {8..15,24..31}
        b.q[0]  = *(const i4*)(Wrow + k0 + 8 * g);
        b.q[1]  = *(const i4*)(Wrow + k0 + 16 + 8 * g);
        // A 16-bit 16x32 layout: g=0 -> K {0..7,16..23}; g=1 -> K {8..15,24..31}
        a0.q[0] = *(const i4*)(Arow0 + k0 + 8 * g);
        a0.q[1] = *(const i4*)(Arow0 + k0 + 16 + 8 * g);
        a1.q[0] = *(const i4*)(Arow1 + k0 + 8 * g);
        a1.q[1] = *(const i4*)(Arow1 + k0 + 16 + 8 * g);
        acc0 = __builtin_amdgcn_wmma_f32_16x16x32_bf16(
                   false, a0.v, false, b.v, (short)0, acc0, false, false);
        acc1 = __builtin_amdgcn_wmma_f32_16x16x32_bf16(
                   false, a1.v, false, b.v, (short)0, acc1, false, false);
    }

    const float bv = p.bias ? p.bias[nt * 16 + r] : 0.0f;
#pragma unroll
    for (int j = 0; j < 8; ++j) {
        // C layout: VGPR j, lanes 0-15 -> M=j, lanes 16-31 -> M=8+j; N = lane%16
        int M0 = j + 8 * g;
        p.C[(size_t)M0        * p.ldc + (nt * 16 + r)] = acc0[j] + bv;
        p.C[(size_t)(16 + M0) * p.ldc + (nt * 16 + r)] = acc1[j] + bv;
    }
}

// single GEMM (logits): grid.x = N/16/WPB
__global__ void gemm_bf16_wmma(GemmArgs p) {
    gemm_body(p, blockIdx.x * WPB + threadIdx.y);
}

// fused pair (GRU input-gates + hidden-gates in one launch): grid.y selects the GEMM
__global__ void gemm_pair_bf16_wmma(GemmArgs p0, GemmArgs p1) {
    const GemmArgs& p = (blockIdx.y == 0) ? p0 : p1;
    gemm_body(p, blockIdx.x * WPB + threadIdx.y);
}

// ---------------- attention + embedding + x0 build (per step) ----------------
// grid = 32 (batch), block = 64 (one thread per source position)
__global__ void attn_embed_kernel(const float* __restrict__ h1,     // [32,512]
                                  const float* __restrict__ enc,    // [32,64,512]
                                  const int* __restrict__ mask,     // [32,64]
                                  const float* __restrict__ emb,    // [V,512]
                                  const int* __restrict__ tgt,      // [32,64]
                                  int step,
                                  float* __restrict__ ctx,          // [32,512]
                                  unsigned short* __restrict__ x0bf,// [32,1024]
                                  float* __restrict__ attn_out) {   // [32,63,64]
    __shared__ float sc[TSRC];
    __shared__ float ae[TSRC];
    const int b = blockIdx.x;
    const int t = threadIdx.x;

    const float* hq   = h1 + b * HH;
    const float* erow = enc + ((size_t)b * TSRC + t) * HH;
    float s = 0.f;
    for (int h = 0; h < HH; ++h) s += erow[h] * hq[h];
    if (mask[b * TSRC + t] == 0) s = -1e9f;
    sc[t] = s;
    __syncthreads();
    float mx = -3.4e38f;
    for (int i = 0; i < TSRC; ++i) mx = fmaxf(mx, sc[i]);
    float e = expf(s - mx);
    ae[t] = e;
    __syncthreads();
    float sum = 0.f;
    for (int i = 0; i < TSRC; ++i) sum += ae[i];
    float a = e / sum;
    attn_out[((size_t)b * LLEN + step) * TSRC + t] = a;
    sc[t] = a;                       // sc reads finished before 2nd barrier
    __syncthreads();

    // ctx[b,h] = sum_t attn[t] * enc[b,t,h]  (threads split the h dimension)
    const float* eb = enc + (size_t)b * TSRC * HH;
    for (int h = t; h < HH; h += TSRC) {
        float c = 0.f;
        for (int tt = 0; tt < TSRC; ++tt) c += sc[tt] * eb[tt * HH + h];
        ctx[b * HH + h] = c;
        x0bf[b * (EE + HH) + EE + h] = f2bf(c);
    }
    // embedding -> first half of x0
    const int tok = tgt[b * 64 + step];
    const float* er2 = emb + (size_t)tok * EE;
    for (int h = t; h < EE; h += TSRC) x0bf[b * (EE + HH) + h] = f2bf(er2[h]);
}

// ---------------- GRU pointwise (layer 0): update h0, emit bf16 copy ----------------
__global__ void gru_point0(const float* __restrict__ gi, const float* __restrict__ gh,
                           float* __restrict__ h, unsigned short* __restrict__ hbf) {
    int idx = blockIdx.x * blockDim.x + threadIdx.x;
    if (idx >= BB * HH) return;
    int b = idx >> 9, j = idx & (HH - 1);
    const float* gib = gi + (size_t)b * G3H;
    const float* ghb = gh + (size_t)b * G3H;
    float rr = 1.f / (1.f + expf(-(gib[j] + ghb[j])));
    float z  = 1.f / (1.f + expf(-(gib[HH + j] + ghb[HH + j])));
    float n  = tanhf(gib[2 * HH + j] + rr * ghb[2 * HH + j]);
    float hp = h[idx];
    float hn = (1.f - z) * n + z * hp;
    h[idx]   = hn;
    hbf[idx] = f2bf(hn);
}

// ---------------- GRU pointwise (layer 1): update h1, build concat_out bf16 ----------
__global__ void gru_point1(const float* __restrict__ gi, const float* __restrict__ gh,
                           float* __restrict__ h, unsigned short* __restrict__ hbf,
                           const float* __restrict__ ctx, unsigned short* __restrict__ cobf) {
    int idx = blockIdx.x * blockDim.x + threadIdx.x;
    if (idx >= BB * HH) return;
    int b = idx >> 9, j = idx & (HH - 1);
    const float* gib = gi + (size_t)b * G3H;
    const float* ghb = gh + (size_t)b * G3H;
    float rr = 1.f / (1.f + expf(-(gib[j] + ghb[j])));
    float z  = 1.f / (1.f + expf(-(gib[HH + j] + ghb[HH + j])));
    float n  = tanhf(gib[2 * HH + j] + rr * ghb[2 * HH + j]);
    float hp = h[idx];
    float hn = (1.f - z) * n + z * hp;
    h[idx]   = hn;
    unsigned short hb = f2bf(hn);
    hbf[idx] = hb;
    cobf[(size_t)b * (2 * HH) + j]      = hb;
    cobf[(size_t)b * (2 * HH) + HH + j] = f2bf(ctx[idx]);
}

extern "C" void kernel_launch(void* const* d_in, const int* in_sizes, int n_in,
                              void* d_out, int out_size, void* d_ws, size_t ws_size,
                              hipStream_t stream) {
    (void)in_sizes; (void)n_in; (void)out_size; (void)ws_size;
    const int*   tgt     = (const int*)d_in[0];
    const float* enc_hid = (const float*)d_in[1];   // [2,32,512]
    const float* enc     = (const float*)d_in[2];   // [32,64,512]
    const int*   mask    = (const int*)d_in[3];     // [32,64]
    const float* emb     = (const float*)d_in[4];   // [V,512]
    const float* W_ih0   = (const float*)d_in[5];   // [1536,1024]
    const float* W_hh0   = (const float*)d_in[6];   // [1536,512]
    const float* b_ih0   = (const float*)d_in[7];
    const float* b_hh0   = (const float*)d_in[8];
    const float* W_ih1   = (const float*)d_in[9];   // [1536,512]
    const float* W_hh1   = (const float*)d_in[10];  // [1536,512]
    const float* b_ih1   = (const float*)d_in[11];
    const float* b_hh1   = (const float*)d_in[12];
    const float* W_out   = (const float*)d_in[13];  // [32000,1024]
    const float* b_out   = (const float*)d_in[14];
    float* out = (float*)d_out;

    // d_out layout (tuple, flat, return order):
    //   outputs [32,63,32000], hidden [2,32,512], all_attn [32,63,64]
    const size_t OUT_HIDDEN = (size_t)BB * LLEN * VV;
    const size_t OUT_ATTN   = OUT_HIDDEN + 2ull * BB * HH;

    // ---- workspace carve-up ----
    char* ws = (char*)d_ws;
    size_t off = 0;
    auto carve = [&](size_t bytes) -> char* {
        char* p = ws + off;
        off = (off + bytes + 255) & ~(size_t)255;
        return p;
    };
    unsigned short* Wout_bf = (unsigned short*)carve((size_t)VV * 1024 * 2);
    unsigned short* Wih0_bf = (unsigned short*)carve((size_t)G3H * 1024 * 2);
    unsigned short* Whh0_bf = (unsigned short*)carve((size_t)G3H * HH * 2);
    unsigned short* Wih1_bf = (unsigned short*)carve((size_t)G3H * HH * 2);
    unsigned short* Whh1_bf = (unsigned short*)carve((size_t)G3H * HH * 2);
    unsigned short* x0bf    = (unsigned short*)carve((size_t)BB * 1024 * 2);
    unsigned short* cobf    = (unsigned short*)carve((size_t)BB * 1024 * 2);
    unsigned short* hbf     = (unsigned short*)carve(2ull * BB * HH * 2); // h0bf, h1bf
    float* gi   = (float*)carve((size_t)BB * G3H * 4);
    float* gh   = (float*)carve((size_t)BB * G3H * 4);
    float* hbuf = (float*)carve(2ull * BB * HH * 4);                      // h0, h1 fp32
    float* ctx  = (float*)carve((size_t)BB * HH * 4);

    unsigned short* h0bf = hbf;
    unsigned short* h1bf = hbf + (size_t)BB * HH;
    float* h0 = hbuf;
    float* h1 = hbuf + (size_t)BB * HH;

    auto cvt = [&](const float* s, unsigned short* d, size_t n) {
        int blocks = (int)((n + 255) / 256);
        cvt_bf16_kernel<<<blocks, 256, 0, stream>>>(s, d, (int)n);
    };

    // one-time per launch: bf16 weight copies + hidden-state init
    cvt(W_out, Wout_bf, (size_t)VV * 1024);
    cvt(W_ih0, Wih0_bf, (size_t)G3H * 1024);
    cvt(W_hh0, Whh0_bf, (size_t)G3H * HH);
    cvt(W_ih1, Wih1_bf, (size_t)G3H * HH);
    cvt(W_hh1, Whh1_bf, (size_t)G3H * HH);
    cvt(enc_hid, hbf, 2ull * BB * HH);
    hipMemcpyAsync(hbuf, enc_hid, 2ull * BB * HH * 4, hipMemcpyDeviceToDevice, stream);

    const dim3 blkGemm(32, WPB);
    const dim3 gGates(G3H / 16 / WPB, 2);   // 24 x 2 (pair kernel)
    const dim3 gLogit(VV / 16 / WPB, 1);    // 500 blocks, 4 waves each
    const int  ptBlocks = (BB * HH + 255) / 256;

    for (int step = 0; step < LLEN; ++step) {
        // 1) attention (uses previous h1), ctx, x0 = [emb(tok), ctx] in bf16
        attn_embed_kernel<<<BB, TSRC, 0, stream>>>(
            h1, enc, mask, emb, tgt, step, ctx, x0bf, out + OUT_ATTN);

        // 2) layer-0 gates: gi = x0 @ W_ih0^T + b_ih0 ; gh = h0 @ W_hh0^T + b_hh0
        {
            GemmArgs pi{ x0bf, 1024, Wih0_bf, 1024, b_ih0, gi, G3H, 1024 };
            GemmArgs ph{ h0bf, HH,   Whh0_bf, HH,   b_hh0, gh, G3H, HH   };
            gemm_pair_bf16_wmma<<<gGates, blkGemm, 0, stream>>>(pi, ph);
        }
        gru_point0<<<ptBlocks, 256, 0, stream>>>(gi, gh, h0, h0bf);

        // 3) layer-1 gates (h0bf fresh; h1bf still previous step's h1)
        {
            GemmArgs pi{ h0bf, HH, Wih1_bf, HH, b_ih1, gi, G3H, HH };
            GemmArgs ph{ h1bf, HH, Whh1_bf, HH, b_hh1, gh, G3H, HH };
            gemm_pair_bf16_wmma<<<gGates, blkGemm, 0, stream>>>(pi, ph);
        }
        gru_point1<<<ptBlocks, 256, 0, stream>>>(gi, gh, h1, h1bf, ctx, cobf);

        // 4) output projection: logits[b, step, :] = [h1, ctx] @ W_out^T + b_out
        {
            GemmArgs pl{ cobf, 1024, Wout_bf, 1024, b_out,
                         out + (size_t)step * VV, (long long)LLEN * VV, 1024 };
            gemm_bf16_wmma<<<gLogit, blkGemm, 0, stream>>>(pl);
        }
    }

    // final hidden state [2,32,512]
    hipMemcpyAsync(out + OUT_HIDDEN, hbuf, 2ull * BB * HH * 4,
                   hipMemcpyDeviceToDevice, stream);
}
